// classifier_stft_6476810682543
// MI455X (gfx1250) — matrix operand order repaired and checked by reading, not compiled, not run on hardware
//
#include <hip/hip_runtime.h>
#include <hip/hip_bf16.h>
#include <math.h>

// ---------------- WMMA types (CDNA5 gfx1250, wave32) ----------------
typedef __attribute__((ext_vector_type(16))) _Float16 v16h;
typedef __attribute__((ext_vector_type(8)))  _Float16 v8h;
typedef __attribute__((ext_vector_type(8)))  float    v8f;

#define PI_F 3.14159265358979323846f
#define NEG_SLOPE 0.01f
#define HFREQ 8192           // MIN_HALF
#define WCOL 6               // C*(HARMONICS+1)
#define FFT_RES_F 0.95367431640625f  // 1/(6.4e-5 * 16384)

// Fused conv tiling
#define TFREQ 64
#define R_C (TFREQ + 14)     // combined rows staged (halo 7 each side)
#define R0   (TFREQ + 12)    // y0 rows (halo 6)
#define R1   (TFREQ + 8)     // y1 rows (halo 4)
#define R2   (TFREQ)         // y2 rows
#define CP0 8                // combined channels padded 7 -> 8
#define N_COMB (R_C * WCOL * CP0)          // 3744
#define N_Y0   (464 * 64)                  // layer0 spatial 456 padded to 464
#define N_Y1   (R1 * WCOL * 96)            // 41472 (432 % 16 == 0)
#define N_Y2   (R2 * WCOL * 128)           // 49152 (384 % 16 == 0)
#define ACT_LDS_BYTES ((N_COMB + N_Y0 + N_Y1 + N_Y2) * 2)   // 248,128
#define WF_LDS_BYTES  (9 * 8 * 512 * 2)                      // 73,728 (max layer)
#define FUSED_LDS_BYTES (ACT_LDS_BYTES + WF_LDS_BYTES)       // 321,856 <= 320KB LDS
#define FUSED_THREADS 512
#define FUSED_WAVES (FUSED_THREADS / 32)

// Pre-packed weight-fragment sizes (halves): KSTEPS * (COUT/16) * 32 * 16
#define WF0_ELEMS (1 * 4 * 512)      //  2048
#define WF1_ELEMS (6 * 6 * 512)      // 18432
#define WF2_ELEMS (9 * 8 * 512)      // 36864

// ====================================================================
// Kernel 1: per-(b,c) mean of 65536 samples
// ====================================================================
__global__ __launch_bounds__(256) void mean_kernel(const float* __restrict__ x,
                                                   float* __restrict__ means) {
    __shared__ float red[256];
    const int bc = blockIdx.x;                     // 0..15
    const float* xs = x + (size_t)bc * 65536;
    float s = 0.f;
    for (int i = threadIdx.x; i < 65536; i += 256) s += xs[i];
    red[threadIdx.x] = s;
    __syncthreads();
    for (int o = 128; o > 0; o >>= 1) {
        if (threadIdx.x < o) red[threadIdx.x] += red[threadIdx.x + o];
        __syncthreads();
    }
    if (threadIdx.x == 0) means[bc] = red[0] * (1.f / 65536.f);
}

// ====================================================================
// Kernel 2: STFT power + harmonic stack + freq avg-pool -> combined f16
// One workgroup per (b, c, frame). Whole half-length complex FFT lives
// in LDS (max 256 KB for N=32768 -- fits the 320 KB WGP LDS).
// ====================================================================
__global__ __launch_bounds__(1024) void stft_kernel(const float* __restrict__ x,
                                                    const float* __restrict__ means,
                                                    _Float16* __restrict__ comb,
                                                    int L, int logN, int frames,
                                                    int pool, int rowBase) {
    extern __shared__ char dyn_smem[];
    const int N = 1 << logN;                       // = L/2
    float* re = (float*)dyn_smem;
    float* im = re + N;

    int idx = blockIdx.x;
    const int fr = idx % frames; idx /= frames;
    const int c = idx & 1;
    const int b = idx >> 1;
    const float mean = means[b * 2 + c];
    const float* xs = x + ((size_t)(b * 2 + c)) * 65536 + (size_t)fr * L;
    const int tid = threadIdx.x;

    // --- load with even/odd pack + bit reversal ---
    for (int i = tid; i < N; i += 1024) {
        const unsigned r = __brev((unsigned)i) >> (32 - logN);
        re[r] = xs[2 * i]     - mean;
        im[r] = xs[2 * i + 1] - mean;
    }
    __syncthreads();

    // --- radix-2 DIT stages, in place ---
    for (int s = 1; s <= logN; ++s) {
        const int m = 1 << s, hm = m >> 1;
        const float astep = -2.f * PI_F / (float)m;
        for (int t = tid; t < (N >> 1); t += 1024) {
            const int pos = t & (hm - 1);
            const int grp = t >> (s - 1);
            const int i1 = grp * m + pos, i2 = i1 + hm;
            float sw, cw;
            __sincosf(astep * (float)pos, &sw, &cw);
            const float r2 = re[i2], q2 = im[i2];
            const float tr = cw * r2 - sw * q2;
            const float ti = cw * q2 + sw * r2;
            const float r1 = re[i1], q1 = im[i1];
            re[i1] = r1 + tr; im[i1] = q1 + ti;
            re[i2] = r1 - tr; im[i2] = q1 - ti;
        }
        __syncthreads();
    }

    // --- real-FFT untangle -> power spectrum (into re[]) ---
    const float invL = 1.f / (float)L;
    for (int k = tid; k <= (N >> 1); k += 1024) {
        if (k == 0) {
            const float v = re[0] + im[0];         // X[0] real
            re[0] = v * v * invL;
        } else {
            const int kn = N - k;
            const float r1 = re[k], q1 = im[k], r2 = re[kn], q2 = im[kn];
            const float xer = 0.5f * (r1 + r2), xei = 0.5f * (q1 - q2);
            const float xor_ = 0.5f * (q1 + q2), xoi = -0.5f * (r1 - r2);
            float sw, cw;
            __sincosf(-PI_F * (float)k / (float)N, &sw, &cw);
            const float twr = cw * xor_ - sw * xoi;
            const float twi = cw * xoi + sw * xor_;
            const float Xr = xer + twr, Xi = xei + twi;      // X[k]
            const float Yr = xer - twr, Yi = xei - twi;      // |X[N-k]| parts
            re[k]  = (Xr * Xr + Xi * Xi) * invL;
            re[kn] = (Yr * Yr + Yi * Yi) * invL;
        }
    }
    __syncthreads();

    // --- harmonic sums + avg pool + store as f16 ---
    const float invp = 1.f / (float)pool;
    const int row = rowBase + fr;
    for (int g = tid; g < HFREQ; g += 1024) {
        float s0 = 0.f, s1 = 0.f, s2 = 0.f;
        for (int t = 0; t < pool; ++t) {
            const int f = pool * g - (pool >> 1) + t;
            if (f >= 0 && f < N) {
                const float P  = re[f];
                const float h1 = P + re[f >> 1];
                const float h3 = h1 + re[f / 3] + re[f >> 2];
                s0 += P; s1 += h1; s2 += h3;
            }
        }
        s0 *= invp;
        s1 *= invp * 0.70710678118654752f;   // /sqrt(2)
        s2 *= invp * 0.5f;                   // /sqrt(4)
        const size_t base = ((((size_t)b * 7 + row) * HFREQ) + g) * WCOL + c * 3;
        comb[base + 0] = (_Float16)s0;
        comb[base + 1] = (_Float16)s1;
        comb[base + 2] = (_Float16)s2;
    }
}

// ====================================================================
// Weight-fragment pre-pack: f32 OIHW -> f16 fragments in WMMA B layout.
// K ordered (kh, ci) over padded CinP; element e of lane holds
// K = k0 + (e&7) + 8*half + 16*(e>>3), N = nt*16 + (lane&15).
// ====================================================================
__global__ __launch_bounds__(256) void wfrag_kernel(const float* __restrict__ W,
                                                    _Float16* __restrict__ out,
                                                    int Cin, int CinP, int Cout) {
    const int NT = Cout >> 4;
    const int KSTEPS = (3 * CinP + 31) >> 5;
    const int total = KSTEPS * NT * 32 * 16;
    for (int i = blockIdx.x * blockDim.x + threadIdx.x; i < total;
         i += gridDim.x * blockDim.x) {
        const int e = i & 15;
        const int lane = (i >> 4) & 31;
        const int rest = i >> 9;
        const int nt = rest % NT;
        const int s = rest / NT;
        const int half = lane >> 4, lr = lane & 15;
        const int k = s * 32 + (e & 7) + half * 8 + (e >> 3) * 16;
        const int n = nt * 16 + lr;
        float v = 0.f;
        if (k < 3 * CinP) {
            const int kh = k / CinP, ci = k - kh * CinP;
            if (ci < Cin) v = W[(n * Cin + ci) * 3 + kh];
        }
        out[i] = (_Float16)v;
    }
}

// ====================================================================
// Async DMA stage of pre-packed B fragments into LDS:
// GLOBAL_LOAD_ASYNC_TO_LDS_B128 (ASYNCcnt) + S_WAIT_ASYNCCNT.
// ====================================================================
__device__ __forceinline__ void stage_wfrag_async(const _Float16* __restrict__ g,
                                                  _Float16* s, int bytes, int tid) {
    const unsigned ldsBase = (unsigned)(uintptr_t)s;
    for (int off = tid * 16; off < bytes; off += FUSED_THREADS * 16) {
        const unsigned long long ga =
            (unsigned long long)(uintptr_t)((const char*)g + off);
        const unsigned la = ldsBase + (unsigned)off;
        asm volatile("global_load_async_to_lds_b128 %0, %1, off"
                     :: "v"(la), "v"(ga) : "memory");
    }
    asm volatile("s_wait_asynccnt 0x0" ::: "memory");
}

// ====================================================================
// One dilated conv layer (K=3 along freq), LDS->LDS implicit GEMM on
// V_WMMA_F32_16X16X32_F16. Each wave owns a fixed out-channel tile nt,
// so the full B operand is preloaded to registers; the inner loop is
// 2x ds_load_b128 (A) + 1 WMMA per K-step. Stores are unguarded
// (dst padded to MT*16 spatial rows).
// ====================================================================
template <int CINP, int COUT, int DIL, int ROWS_OUT>
__device__ __forceinline__ void conv_layer_t(const _Float16* __restrict__ src,
                                             _Float16* __restrict__ dst,
                                             const _Float16* __restrict__ wfrag,
                                             const float* __restrict__ bias,
                                             int g_base, int tid) {
    constexpr int M = ROWS_OUT * WCOL;
    constexpr int MT = (M + 15) >> 4;
    constexpr int NT = COUT >> 4;
    constexpr int KTOT = 3 * CINP;
    constexpr int KSTEPS = (KTOT + 31) >> 5;
    const int wave = tid >> 5;
    const int lane = tid & 31;
    const int half = lane >> 4;
    const int lr = lane & 15;

    // Fixed out-channel tile per wave; partition m-tiles among its waves.
    const int nt = wave % NT;
    const int mt0 = wave / NT;
    const int mtStride = (FUSED_WAVES - 1 - nt) / NT + 1;

    // ---- preload B fragments (from LDS-staged wfrag) into registers ----
    v16h breg[KSTEPS];
#pragma unroll
    for (int s = 0; s < KSTEPS; ++s) {
        const v8h* bp = (const v8h*)(wfrag + (((size_t)s * NT + nt) * 32 + lane) * 16);
        union { v16h v; v8h h[2]; } bf;
        bf.h[0] = bp[0];
        bf.h[1] = bp[1];
        breg[s] = bf.v;
    }
    const int co = nt * 16 + lr;
    const float bia = bias[co];

    for (int mt = mt0; mt < MT; mt += mtStride) {
        const int m = mt * 16 + lr;
        const int mld = (m < M) ? m : (M - 1);       // clamp loads only
        const int hl = mld / WCOL, w = mld - hl * WCOL;
        const int abase = (hl * WCOL + w) * CINP;
        v8f acc = {};
#pragma unroll
        for (int s = 0; s < KSTEPS; ++s) {
            const int kb1 = s * 32 + half * 8;
            const int kb2 = s * 32 + 16 + half * 8;
            const int kh1 = kb1 / CINP, ci1 = kb1 - kh1 * CINP;
            const int kh2 = kb2 / CINP, ci2 = kb2 - kh2 * CINP;
            union { v16h v; v8h h[2]; } a;
            a.h[0] = *(const v8h*)(src + abase + kh1 * (DIL * WCOL * CINP) + ci1);
            if ((KTOT % 32 == 0) || (kb2 < KTOT)) {
                a.h[1] = *(const v8h*)(src + abase + kh2 * (DIL * WCOL * CINP) + ci2);
            } else {
                v8h z = {};
                a.h[1] = z;
            }
            acc = __builtin_amdgcn_wmma_f32_16x16x32_f16(
                false, a.v, false, breg[s], (short)0, acc, false, false);
        }
        // ---- epilogue: bias + leaky, zero rows outside [0,8192) ----
#pragma unroll
        for (int r = 0; r < 8; ++r) {
            const int ms = mt * 16 + half * 8 + r;
            const int hls = ms / WCOL;
            const int ws_ = ms - hls * WCOL;
            float v = acc[r] + bia;
            v = (v >= 0.f) ? v : NEG_SLOPE * v;
            const int gh = g_base + hls;
            if ((unsigned)gh >= (unsigned)HFREQ) v = 0.f;
            dst[(hls * WCOL + ws_) * COUT + co] = (_Float16)v;
        }
    }
}

// ====================================================================
// Kernel 3: fully fused conv stack, one block per (batch, 64-freq tile).
// Activations + current layer's B fragments all live in LDS (~322 KB).
// ====================================================================
__global__ __launch_bounds__(FUSED_THREADS) void fused_conv_kernel(
        const _Float16* __restrict__ comb,
        const _Float16* __restrict__ wf0, const float* __restrict__ b0,
        const _Float16* __restrict__ wf1, const float* __restrict__ b1,
        const _Float16* __restrict__ wf2, const float* __restrict__ b2,
        const float* __restrict__ w3, const float* __restrict__ b3,
        float* __restrict__ y3) {
    extern __shared__ char dyn_smem[];
    _Float16* sComb = (_Float16*)dyn_smem;
    _Float16* sY0 = sComb + N_COMB;
    _Float16* sY1 = sY0 + N_Y0;
    _Float16* sY2 = sY1 + N_Y1;
    _Float16* sWf = (_Float16*)(dyn_smem + ACT_LDS_BYTES);

    const int tid = threadIdx.x;
    const int b = blockIdx.y;
    const int h0 = blockIdx.x * TFREQ;

    // Warm L2/WGP$ for the next tile's combined region (global_prefetch_b8).
    if (tid == 0) {
        const int hn = (h0 + TFREQ < HFREQ) ? h0 + TFREQ : h0;
        __builtin_prefetch((const void*)(comb + (((size_t)b * 7) * HFREQ + hn) * WCOL), 0, 1);
    }

    // ---- async-DMA layer0 B fragments while VALU stages combined ----
    stage_wfrag_async(wf0, sWf, WF0_ELEMS * 2, tid);

    // ---- stage combined tile (halo 7, channel-padded 7->8, zero OOB) ----
    for (int e = tid; e < R_C * WCOL * CP0; e += FUSED_THREADS) {
        const int hc = e / (WCOL * CP0);
        const int rem = e - hc * (WCOL * CP0);
        const int w = rem / CP0;
        const int ch = rem - w * CP0;
        const int gh = h0 - 7 + hc;
        _Float16 v = (_Float16)0.f;
        if (ch < 7 && gh >= 0 && gh < HFREQ)
            v = comb[(((size_t)b * 7 + ch) * HFREQ + gh) * WCOL + w];
        sComb[(hc * WCOL + w) * CP0 + ch] = v;
    }
    __syncthreads();

    // ---- conv0: 7(->8) -> 64, dil 1 ----
    conv_layer_t<CP0, 64, 1, R0>(sComb, sY0, sWf, b0, h0 - 6, tid);
    __syncthreads();
    stage_wfrag_async(wf1, sWf, WF1_ELEMS * 2, tid);
    __syncthreads();
    // ---- conv1: 64 -> 96, dil 2 ----
    conv_layer_t<64, 96, 2, R1>(sY0, sY1, sWf, b1, h0 - 4, tid);
    __syncthreads();
    stage_wfrag_async(wf2, sWf, WF2_ELEMS * 2, tid);
    __syncthreads();
    // ---- conv2: 96 -> 128, dil 4 ----
    conv_layer_t<96, 128, 4, R2>(sY1, sY2, sWf, b2, h0, tid);
    __syncthreads();

    // ---- conv3: 1x1, 128 -> 1 (VALU dot, vectorized LDS reads) ----
    const float bias3 = b3[0];
    for (int s = tid; s < TFREQ * WCOL; s += FUSED_THREADS) {
        float acc = bias3;
        const v8h* rp = (const v8h*)(sY2 + s * 128);
#pragma unroll
        for (int cb = 0; cb < 16; ++cb) {
            const v8h hv = rp[cb];
#pragma unroll
            for (int j = 0; j < 8; ++j) acc += (float)hv[j] * w3[cb * 8 + j];
        }
        const int hl = s / WCOL, w = s - hl * WCOL;
        y3[((size_t)b * HFREQ + (h0 + hl)) * WCOL + w] = acc;
    }
}

// ====================================================================
// Kernel 4: mask edges, per-batch max/argmax (first-max tiebreak),
// affine head + period -> 3 floats per batch.
// ====================================================================
__global__ __launch_bounds__(256) void finalize_kernel(const float* __restrict__ y3,
                                                       const float* __restrict__ fw,
                                                       const float* __restrict__ fb,
                                                       float* __restrict__ out) {
    __shared__ float bv[256];
    __shared__ int   bi[256];
    const int b = blockIdx.x;
    const int tx = threadIdx.x;
    const float* ys = y3 + (size_t)b * HFREQ * WCOL;
    float best = -3.4e38f;
    int besti = 0;
    for (int i = tx; i < HFREQ * WCOL; i += 256) {
        const int h = i / WCOL;
        const float v = (h < 50 || h >= HFREQ - 50) ? -100.f : ys[i];
        if (v > best) { best = v; besti = i; }
    }
    bv[tx] = best; bi[tx] = besti;
    __syncthreads();
    for (int o = 128; o > 0; o >>= 1) {
        if (tx < o) {
            if (bv[tx + o] > bv[tx] || (bv[tx + o] == bv[tx] && bi[tx + o] < bi[tx])) {
                bv[tx] = bv[tx + o]; bi[tx] = bi[tx + o];
            }
        }
        __syncthreads();
    }
    if (tx == 0) {
        const float mx = bv[0];
        const int h = bi[0] / WCOL;
        out[b * 3 + 0] = mx * fw[0] + fb[0];
        out[b * 3 + 1] = mx * fw[1] + fb[1];
        out[b * 3 + 2] = 1.f / ((float)h * FFT_RES_F + 1e-7f);
    }
}

// ====================================================================
// Host launcher
// ====================================================================
extern "C" void kernel_launch(void* const* d_in, const int* in_sizes, int n_in,
                              void* d_out, int out_size, void* d_ws, size_t ws_size,
                              hipStream_t stream) {
    (void)in_sizes; (void)n_in; (void)out_size; (void)ws_size;
    const float* x  = (const float*)d_in[0];
    const float* w0 = (const float*)d_in[1];
    const float* b0 = (const float*)d_in[2];
    const float* w1 = (const float*)d_in[3];
    const float* b1 = (const float*)d_in[4];
    const float* w2 = (const float*)d_in[5];
    const float* b2 = (const float*)d_in[6];
    const float* w3 = (const float*)d_in[7];
    const float* b3 = (const float*)d_in[8];
    const float* fw = (const float*)d_in[9];
    const float* fb = (const float*)d_in[10];
    float* out = (float*)d_out;

    // Workspace carve:
    //   means(256B) | combined f16 | y3 f32 | wf0 | wf1 | wf2
    char* ws = (char*)d_ws;
    float* means = (float*)ws;
    _Float16* comb = (_Float16*)(ws + 256);
    const size_t combBytes = (size_t)8 * 7 * HFREQ * WCOL * 2;   // 5,505,024
    float* y3 = (float*)(ws + 256 + combBytes);
    const size_t y3Bytes = (size_t)8 * HFREQ * WCOL * 4;         // 1,572,864
    _Float16* wf0 = (_Float16*)(ws + 256 + combBytes + y3Bytes);
    _Float16* wf1 = wf0 + WF0_ELEMS;
    _Float16* wf2 = wf1 + WF1_ELEMS;

    mean_kernel<<<16, 256, 0, stream>>>(x, means);

    // Pre-pack conv weights into WMMA B fragments (f16).
    wfrag_kernel<<<8,  256, 0, stream>>>(w0, wf0, 7,  CP0, 64);
    wfrag_kernel<<<32, 256, 0, stream>>>(w1, wf1, 64, 64,  96);
    wfrag_kernel<<<64, 256, 0, stream>>>(w2, wf2, 96, 96,  128);

    // j=0: L=65536, N=32768, pool=4, row 0        (LDS 256 KB)
    stft_kernel<<<16, 1024, 2 * 32768 * sizeof(float), stream>>>(
        x, means, comb, 65536, 15, 1, 4, 0);
    // j=1: L=32768, N=16384, pool=2, rows 1..2    (LDS 128 KB)
    stft_kernel<<<32, 1024, 2 * 16384 * sizeof(float), stream>>>(
        x, means, comb, 32768, 14, 2, 2, 1);
    // j=2: L=16384, N=8192,  pool=1, rows 3..6    (LDS 64 KB)
    stft_kernel<<<64, 1024, 2 * 8192 * sizeof(float), stream>>>(
        x, means, comb, 16384, 13, 4, 1, 3);

    fused_conv_kernel<<<dim3(HFREQ / TFREQ, 8), FUSED_THREADS, FUSED_LDS_BYTES, stream>>>(
        comb, wf0, b0, wf1, b1, wf2, b2, w3, b3, y3);

    finalize_kernel<<<8, 256, 0, stream>>>(y3, fw, fb, out);
}
